// STA_LSTM_72928544686310
// MI455X (gfx1250) — matrix operand order repaired and checked by reading, not compiled, not run on hardware
//
#include <hip/hip_runtime.h>
#include <hip/hip_bf16.h>
#include <math.h>
#include <stdint.h>

typedef _Float16 v16h __attribute__((ext_vector_type(16)));
typedef _Float16 v8h  __attribute__((ext_vector_type(8)));
typedef _Float16 v4h  __attribute__((ext_vector_type(4)));
typedef float    v8f  __attribute__((ext_vector_type(8)));
typedef int      v4i  __attribute__((ext_vector_type(4)));

constexpr int NB = 4096;   // batch
constexpr int S  = 16;
constexpr int D  = 256;
constexpr int HS = 512;
constexpr int HT = 512;

// ---------------- WMMA tiled GEMM ----------------
// Block: 256 threads (8 waves), tile 128(M) x 128(N), K stepped by 32.
// Wave tile: 32(M) x 64(N) -> 8 WMMAs per K-step.
constexpr int TM = 128, TN = 128, TK = 32;
constexpr int LST  = 40;        // LDS row stride in halves (80B rows, 16B aligned)
constexpr int BUFH = 128 * LST; // halves per LDS buffer (10,240B)

// ---- CDNA5 async global->LDS (ASYNCcnt) with portable fallback ----
#if __has_builtin(__builtin_amdgcn_global_load_async_to_lds_b128)
#define USE_ASYNC_LDS 1
#else
#define USE_ASYNC_LDS 0
#endif

__device__ inline void async_copy_b128(const _Float16* gsrc, _Float16* ldst) {
#if USE_ASYNC_LDS
    // signature (probe-learned): (v4i AS1*, v4i AS3*, imm offset, imm cpol)
    __builtin_amdgcn_global_load_async_to_lds_b128(
        (__attribute__((address_space(1))) v4i*)(uintptr_t)gsrc,
        (__attribute__((address_space(3))) v4i*)(uint32_t)(uintptr_t)ldst,
        0, 0);
#else
    *(v8h*)ldst = *(const v8h*)gsrc;
#endif
}

__device__ inline void async_wait0() {
#if USE_ASYNC_LDS
#if __has_builtin(__builtin_amdgcn_s_wait_asynccnt)
    __builtin_amdgcn_s_wait_asynccnt(0);
#else
    asm volatile("s_wait_asynccnt 0x0" ::: "memory");
#endif
#endif
}

__device__ inline v16h load_a_frag(const _Float16* As, int rowBase, int lane) {
    // 16-bit A 16x32 layout: lanes 0-15 -> K[0..7],K[16..23]; lanes 16-31 -> K[8..15],K[24..31]
    int m = lane & 15, hi = lane >> 4;
    const _Float16* p = As + (rowBase + m) * LST + hi * 8;
    v8h lo = *(const v8h*)p;
    v8h hh = *(const v8h*)(p + 16);
    v16h r;
#pragma unroll
    for (int j = 0; j < 8; ++j) { r[j] = lo[j]; r[j + 8] = hh[j]; }
    return r;
}

__device__ inline v16h load_b_frag(const _Float16* Bs, int colBase, int lane) {
    // B 32x16 layout: lanes 0-15 -> K[0..7],K[16..23]; lanes 16-31 -> K[8..15],K[24..31].
    // Bs is N-major: Bs[n][k].
    int n = lane & 15, kh = (lane >> 4) * 8;
    const _Float16* p = Bs + (colBase + n) * LST + kh;
    v8h lo = *(const v8h*)p;
    v8h hh = *(const v8h*)(p + 16);
    v16h r;
#pragma unroll
    for (int j = 0; j < 8; ++j) { r[j] = lo[j]; r[j + 8] = hh[j]; }
    return r;
}

// FULL path: M%128==0, N%128==0, K%32==0. A fp32; WT = f16 weights, N-major
// (WT[n*ldw + k]). Double-buffered LDS, one barrier per K-step, async B staging.
__global__ __launch_bounds__(256) void gemm_wmma_full(
    const float* __restrict__ A, int lda,
    const _Float16* __restrict__ WT, int ldw,
    const float* __restrict__ bias,
    float* __restrict__ C, int ldc,
    int M, int N, int K, int accumulate)
{
    __shared__ __align__(16) _Float16 AsAll[2 * BUFH];
    __shared__ __align__(16) _Float16 BsAll[2 * BUFH];

    const int tid  = threadIdx.x;
    const int lane = tid & 31;
    const int wave = tid >> 5;
    const int wm   = wave & 3;   // 4 M-blocks of 32
    const int wn   = wave >> 2;  // 2 N-blocks of 64
    const int row0 = blockIdx.x * TM;
    const int col0 = blockIdx.y * TN;

    v8f acc[2][4];
#pragma unroll
    for (int i = 0; i < 2; ++i)
#pragma unroll
        for (int j = 0; j < 4; ++j)
#pragma unroll
            for (int r = 0; r < 8; ++r) acc[i][j][r] = 0.f;

    float4 regA[4];  // 128x32 A tile: 1024 float4 / 256 thr = 4 each
    auto loadAreg = [&](int k0) {
#pragma unroll
        for (int i = 0; i < 4; ++i) {
            int iv = i * 256 + tid;
            int r = iv >> 3, kq = iv & 7;
            regA[i] = *(const float4*)(A + (size_t)(row0 + r) * lda + k0 + kq * 4);
        }
    };
    auto stageA = [&](int buf) {
#pragma unroll
        for (int i = 0; i < 4; ++i) {
            int iv = i * 256 + tid;
            int r = iv >> 3, kq = iv & 7;
            v4h h;
            h[0] = (_Float16)regA[i].x; h[1] = (_Float16)regA[i].y;
            h[2] = (_Float16)regA[i].z; h[3] = (_Float16)regA[i].w;
            *(v4h*)(AsAll + buf * BUFH + r * LST + kq * 4) = h;
        }
    };
    const _Float16* Wbase = WT + (size_t)col0 * ldw;
    auto asyncB = [&](int buf, int k0) {
        // 128 n-rows x 32 k halves = 512 x 16B chunks / 256 thr = 2 each
#pragma unroll
        for (int i = 0; i < 2; ++i) {
            int iv = i * 256 + tid;
            int n = iv & 127, kq = iv >> 7;
            const _Float16* src = Wbase + (size_t)n * ldw + k0 + kq * 8;
            _Float16* dst = BsAll + buf * BUFH + n * LST + kq * 8;
            async_copy_b128(src, dst);
        }
    };

    loadAreg(0);
    asyncB(0, 0);
    int cur = 0;
    for (int k0 = 0; k0 < K; k0 += TK, cur ^= 1) {
        stageA(cur);            // implicit s_wait_loadcnt on regA
        async_wait0();          // B tile resident in LDS
        __syncthreads();        // single barrier per K-step
        if (k0 + TK < K) {      // prefetch next tile into other buffer
            loadAreg(k0 + TK);
            asyncB(cur ^ 1, k0 + TK);
        }
        const _Float16* Asb = AsAll + cur * BUFH;
        const _Float16* Bsb = BsAll + cur * BUFH;
        v16h a0 = load_a_frag(Asb, wm * 32,      lane);
        v16h a1 = load_a_frag(Asb, wm * 32 + 16, lane);
#pragma unroll
        for (int j = 0; j < 4; ++j) {
            v16h b = load_b_frag(Bsb, wn * 64 + j * 16, lane);
            acc[0][j] = __builtin_amdgcn_wmma_f32_16x16x32_f16(false, a0, false, b, (short)0, acc[0][j], false, false);
            acc[1][j] = __builtin_amdgcn_wmma_f32_16x16x32_f16(false, a1, false, b, (short)0, acc[1][j], false, false);
        }
    }

    const int n  = lane & 15;
    const int mo = (lane >> 4) * 8;
#pragma unroll
    for (int i = 0; i < 2; ++i)
#pragma unroll
        for (int j = 0; j < 4; ++j) {
            int col = col0 + wn * 64 + j * 16 + n;
#pragma unroll
            for (int r = 0; r < 8; ++r) {
                int row = row0 + wm * 32 + i * 16 + mo + r;
                float v = acc[i][j][r] + (bias ? bias[col] : 0.f);
                size_t idx = (size_t)row * ldc + col;
                if (accumulate) v += C[idx];
                C[idx] = v;
            }
        }
}

// Guarded path (tiny N / ragged K): fp32 W, clamped unconditional loads + select.
__global__ __launch_bounds__(256) void gemm_wmma_guard(
    const float* __restrict__ A, int lda,
    const float* __restrict__ W, int ldb,
    const float* __restrict__ bias,
    float* __restrict__ C, int ldc,
    int M, int N, int K, int accumulate)
{
    __shared__ __align__(16) _Float16 As[BUFH];
    __shared__ __align__(16) _Float16 Bs[BUFH];

    const int tid  = threadIdx.x;
    const int lane = tid & 31;
    const int wave = tid >> 5;
    const int wm   = wave & 3;
    const int wn   = wave >> 2;
    const int row0 = blockIdx.x * TM;
    const int col0 = blockIdx.y * TN;

    v8f acc[2][4];
#pragma unroll
    for (int i = 0; i < 2; ++i)
#pragma unroll
        for (int j = 0; j < 4; ++j)
#pragma unroll
            for (int r = 0; r < 8; ++r) acc[i][j][r] = 0.f;

    for (int k0 = 0; k0 < K; k0 += TK) {
#pragma unroll
        for (int i = 0; i < 16; ++i) {            // 128x32 A elements
            int iv = i * 256 + tid;
            int r = iv >> 5, kk = iv & 31;
            int gr = row0 + r, gk = k0 + kk;
            int grc = min(gr, M - 1), gkc = min(gk, K - 1);
            float v = A[(size_t)grc * lda + gkc];
            As[r * LST + kk] = (_Float16)((gr < M && gk < K) ? v : 0.f);
        }
#pragma unroll
        for (int i = 0; i < 16; ++i) {            // 32x128 B elements
            int iv = i * 256 + tid;
            int n = iv & 127, kk = iv >> 7;
            int gn = col0 + n, gk = k0 + kk;
            int gnc = min(gn, N - 1), gkc = min(gk, K - 1);
            float v = W[(size_t)gkc * ldb + gnc];
            Bs[n * LST + kk] = (_Float16)((gn < N && gk < K) ? v : 0.f);
        }
        __syncthreads();

        v16h a0 = load_a_frag(As, wm * 32,      lane);
        v16h a1 = load_a_frag(As, wm * 32 + 16, lane);
#pragma unroll
        for (int j = 0; j < 4; ++j) {
            v16h b = load_b_frag(Bs, wn * 64 + j * 16, lane);
            acc[0][j] = __builtin_amdgcn_wmma_f32_16x16x32_f16(false, a0, false, b, (short)0, acc[0][j], false, false);
            acc[1][j] = __builtin_amdgcn_wmma_f32_16x16x32_f16(false, a1, false, b, (short)0, acc[1][j], false, false);
        }
        __syncthreads();
    }

    const int n  = lane & 15;
    const int mo = (lane >> 4) * 8;
#pragma unroll
    for (int i = 0; i < 2; ++i)
#pragma unroll
        for (int j = 0; j < 4; ++j) {
            int col = col0 + wn * 64 + j * 16 + n;
#pragma unroll
            for (int r = 0; r < 8; ++r) {
                int row = row0 + wm * 32 + i * 16 + mo + r;
                if (row < M && col < N) {
                    float v = acc[i][j][r] + (bias ? bias[col] : 0.f);
                    size_t idx = (size_t)row * ldc + col;
                    if (accumulate) v += C[idx];
                    C[idx] = v;
                }
            }
        }
}

// out[n*K + k] = (f16) in[k*N + n]   (one-time weight transpose+convert)
__global__ void transpose_to_f16(const float* __restrict__ in, _Float16* __restrict__ out,
                                 int K, int N) {
    int idx = blockIdx.x * blockDim.x + threadIdx.x;
    if (idx >= K * N) return;
    int k = idx / N, n = idx % N;
    out[(size_t)n * K + k] = (_Float16)in[idx];
}

// ---------------- elementwise kernels ----------------
__device__ inline float sigm(float x) { return 1.f / (1.f + __expf(-x)); }

__global__ void ew_tanh(float* p, int n) {
    int i = blockIdx.x * blockDim.x + threadIdx.x;
    if (i < n) p[i] = tanhf(p[i]);
}

__global__ void zero_f32(float* p, int n) {
    int i = blockIdx.x * blockDim.x + threadIdx.x;
    if (i < n) p[i] = 0.f;
}

// alpha = softmax(a, axis=1); xt = alpha * x_t  (rows of length D=256, wave per row)
__global__ void sa_softmax_scale(const float* __restrict__ a,
                                 const float* __restrict__ X,
                                 float* __restrict__ xt, int t)
{
    int wave = threadIdx.x >> 5, lane = threadIdx.x & 31;
    int b = blockIdx.x * 8 + wave;
    if (b >= NB) return;
    const float* arow = a + (size_t)b * D;
    float v[8];
    float mx = -INFINITY;
#pragma unroll
    for (int i = 0; i < 8; ++i) { v[i] = arow[lane + i * 32]; mx = fmaxf(mx, v[i]); }
#pragma unroll
    for (int off = 16; off; off >>= 1) mx = fmaxf(mx, __shfl_xor(mx, off, 32));
    float s = 0.f;
#pragma unroll
    for (int i = 0; i < 8; ++i) { v[i] = __expf(v[i] - mx); s += v[i]; }
#pragma unroll
    for (int off = 16; off; off >>= 1) s += __shfl_xor(s, off, 32);
    float inv = 1.f / s;
    const float* xrow = X + (size_t)b * S * D + (size_t)t * D;
    float* orow = xt + (size_t)b * D;
#pragma unroll
    for (int i = 0; i < 8; ++i) {
        int c = lane + i * 32;
        orow[c] = v[i] * inv * xrow[c];
    }
}

// in-place softmax over rows of length 16, thread per row
__global__ void ta_softmax16(float* __restrict__ beta) {
    int b = blockIdx.x * blockDim.x + threadIdx.x;
    if (b >= NB) return;
    float* r = beta + (size_t)b * S;
    float mx = -INFINITY;
#pragma unroll
    for (int i = 0; i < S; ++i) mx = fmaxf(mx, r[i]);
    float s = 0.f;
    float e[S];
#pragma unroll
    for (int i = 0; i < S; ++i) { e[i] = __expf(r[i] - mx); s += e[i]; }
    float inv = 1.f / s;
#pragma unroll
    for (int i = 0; i < S; ++i) r[i] = e[i] * inv;
}

// h_ctx[b,h] = sum_s beta[b,s] * hseq[s][b,h]
__global__ void ta_context(const float* __restrict__ beta,
                           const float* __restrict__ hseq,
                           float* __restrict__ hctx)
{
    int idx = blockIdx.x * blockDim.x + threadIdx.x;
    if (idx >= NB * HS) return;
    int b = idx >> 9, h = idx & (HS - 1);
    float s = 0.f;
#pragma unroll
    for (int t = 0; t < S; ++t)
        s += beta[(size_t)b * S + t] * hseq[(size_t)t * NB * HS + (size_t)b * HS + h];
    hctx[idx] = s;
}

// LSTM cell: gates [B,4H], c in-place, h_out written; optional y*wy rank-1 term
__global__ void lstm_update(const float* __restrict__ gates,
                            float* __restrict__ c,
                            float* __restrict__ h_out,
                            const float* __restrict__ y,
                            const float* __restrict__ wy)
{
    int idx = blockIdx.x * blockDim.x + threadIdx.x;
    if (idx >= NB * HS) return;
    int b = idx >> 9, j = idx & (HS - 1);
    const float* g = gates + (size_t)b * (4 * HS);
    float yv = y ? y[b] : 0.f;
    float gi = g[j]            + (wy ? yv * wy[j]            : 0.f);
    float gf = g[HS + j]       + (wy ? yv * wy[HS + j]       : 0.f);
    float gg = g[2 * HS + j]   + (wy ? yv * wy[2 * HS + j]   : 0.f);
    float go = g[3 * HS + j]   + (wy ? yv * wy[3 * HS + j]   : 0.f);
    float iv = sigm(gi), fv = sigm(gf), gv = tanhf(gg), ov = sigm(go);
    float cn = fv * c[idx] + iv * gv;
    c[idx] = cn;
    h_out[idx] = ov * tanhf(cn);
}

// y[b] = dot(lh[b,:], fc_W[0,:]) + fc_b[0]   (wave per row of 512)
__global__ void ta_output(const float* __restrict__ lh,
                          const float* __restrict__ fcW,
                          const float* __restrict__ fcb,
                          float* __restrict__ y)
{
    int wave = threadIdx.x >> 5, lane = threadIdx.x & 31;
    int b = blockIdx.x * 8 + wave;
    if (b >= NB) return;
    const float* r = lh + (size_t)b * HT;
    float s = 0.f;
#pragma unroll
    for (int i = 0; i < 16; ++i) {
        int c = lane + i * 32;
        s += r[c] * fcW[c];
    }
#pragma unroll
    for (int off = 16; off; off >>= 1) s += __shfl_xor(s, off, 32);
    if (lane == 0) y[b] = s + fcb[0];
}

__global__ void init_y0(const float* __restrict__ X, float* __restrict__ y) {
    int b = blockIdx.x * blockDim.x + threadIdx.x;
    if (b < NB) y[b] = X[(size_t)b * S * D + (D - 1)];
}

__global__ void copy_out(const float* __restrict__ y, float* __restrict__ out) {
    int b = blockIdx.x * blockDim.x + threadIdx.x;
    if (b < NB) out[b] = y[b];
}

// ---------------- host driver ----------------
static inline void launch_full(const float* A, int lda, const _Float16* WT, int ldw,
                               const float* bias, float* C, int ldc,
                               int M, int N, int K, int acc, hipStream_t s)
{
    dim3 grid(M / TM, N / TN);
    gemm_wmma_full<<<grid, 256, 0, s>>>(A, lda, WT, ldw, bias, C, ldc, M, N, K, acc);
}

static inline void launch_guard(const float* A, int lda, const float* W, int ldb,
                                const float* bias, float* C, int ldc,
                                int M, int N, int K, int acc, hipStream_t s)
{
    dim3 grid((M + TM - 1) / TM, (N + TN - 1) / TN);
    gemm_wmma_guard<<<grid, 256, 0, s>>>(A, lda, W, ldb, bias, C, ldc, M, N, K, acc);
}

extern "C" void kernel_launch(void* const* d_in, const int* in_sizes, int n_in,
                              void* d_out, int out_size, void* d_ws, size_t ws_size,
                              hipStream_t stream)
{
    (void)in_sizes; (void)n_in; (void)out_size; (void)ws_size;
    const float* X     = (const float*)d_in[0];
    const float* sa_W  = (const float*)d_in[1];
    const float* sa_U  = (const float*)d_in[2];
    const float* sa_b  = (const float*)d_in[3];
    const float* sa_Wa = (const float*)d_in[4];
    const float* sa_Ua = (const float*)d_in[5];
    const float* sa_ba = (const float*)d_in[6];
    const float* sa_Va = (const float*)d_in[7];
    const float* ta_Wa = (const float*)d_in[8];
    // d_in[9] = ta_Ua multiplies s0 == 0 in the reference -> term vanishes
    const float* ta_ba = (const float*)d_in[10];
    const float* ta_Va = (const float*)d_in[11];
    const float* ta_W  = (const float*)d_in[12];
    const float* ta_U  = (const float*)d_in[13];
    const float* ta_b  = (const float*)d_in[14];
    const float* ta_Wy = (const float*)d_in[15];
    const float* fc_W  = (const float*)d_in[16];
    const float* fc_b  = (const float*)d_in[17];
    float* out = (float*)d_out;

    // ---- workspace carve: fp32 region, then f16 weight region ----
    float* p = (float*)d_ws;
    auto take = [&](size_t n) { float* q = p; p += n; return q; };
    float* c_sa  = take((size_t)NB * HS);
    float* hzero = take((size_t)NB * HS);
    float* hseq  = take((size_t)S * NB * HS);
    float* gates = take((size_t)NB * 4 * HS);
    float* t1    = take((size_t)NB * D);
    float* t2    = take((size_t)NB * D);
    float* xt    = take((size_t)NB * D);
    float* bpre  = take((size_t)NB * S);
    float* beta  = take((size_t)NB * S);
    float* hctx  = take((size_t)NB * HT);
    float* lh    = take((size_t)NB * HT);
    float* lc    = take((size_t)NB * HT);
    float* yb    = take((size_t)NB);

    _Float16* hp = (_Float16*)p;
    auto takeh = [&](size_t n) { _Float16* q = hp; hp += n; return q; };
    _Float16* sa_WaT = takeh((size_t)D * D);            // [256 n][256 k]
    _Float16* sa_UaT = takeh((size_t)2 * HS * D);       // [256 n][1024 k]
    _Float16* sa_VaT = takeh((size_t)D * D);
    _Float16* sa_WT  = takeh((size_t)D * 4 * HS);       // [2048 n][256 k]
    _Float16* sa_UT  = takeh((size_t)HS * 4 * HS);      // [2048 n][512 k]
    _Float16* ta_WT  = takeh((size_t)HS * 4 * HT);
    _Float16* ta_UT  = takeh((size_t)HT * 4 * HT);

    const int EW = 256;
    auto tr = [&](const float* w, _Float16* wt, int K, int N) {
        transpose_to_f16<<<(K * N + EW - 1) / EW, EW, 0, stream>>>(w, wt, K, N);
    };
    tr(sa_Wa, sa_WaT, D, D);
    tr(sa_Ua, sa_UaT, 2 * HS, D);
    tr(sa_Va, sa_VaT, D, D);
    tr(sa_W,  sa_WT,  D,  4 * HS);
    tr(sa_U,  sa_UT,  HS, 4 * HS);
    tr(ta_W,  ta_WT,  HS, 4 * HT);
    tr(ta_U,  ta_UT,  HT, 4 * HT);

    int nBH = NB * HS;
    zero_f32<<<(nBH + EW - 1) / EW, EW, 0, stream>>>(c_sa,  nBH);
    zero_f32<<<(nBH + EW - 1) / EW, EW, 0, stream>>>(hzero, nBH);
    zero_f32<<<(nBH + EW - 1) / EW, EW, 0, stream>>>(lh,    nBH);
    zero_f32<<<(nBH + EW - 1) / EW, EW, 0, stream>>>(lc,    nBH);
    init_y0<<<(NB + EW - 1) / EW, EW, 0, stream>>>(X, yb);

    // -------- spatial-attention LSTM scan --------
    for (int t = 0; t < S; ++t) {
        const float* hprev = (t == 0) ? hzero : hseq + (size_t)(t - 1) * NB * HS;
        // pre = x_t @ sa_Wa + h @ sa_Ua[:HS] + c @ sa_Ua[HS:] + sa_ba
        launch_full(X + t * D, S * D, sa_WaT, D,           sa_ba,   t1, D, NB, D, D,  0, stream);
        launch_full(hprev, HS, sa_UaT,       2 * HS,       nullptr, t1, D, NB, D, HS, 1, stream);
        launch_full(c_sa,  HS, sa_UaT + HS,  2 * HS,       nullptr, t1, D, NB, D, HS, 1, stream);
        ew_tanh<<<(NB * D + EW - 1) / EW, EW, 0, stream>>>(t1, NB * D);
        // a_t = tanh(pre) @ sa_Va ; alpha = softmax(a_t); xt = alpha * x_t
        launch_full(t1, D, sa_VaT, D, nullptr, t2, D, NB, D, D, 0, stream);
        sa_softmax_scale<<<NB / 8, 256, 0, stream>>>(t2, X, xt, t);
        // gates = xt @ sa_W + h @ sa_U + sa_b ; LSTM update
        launch_full(xt,    D,  sa_WT, D,  sa_b,    gates, 4 * HS, NB, 4 * HS, D,  0, stream);
        launch_full(hprev, HS, sa_UT, HS, nullptr, gates, 4 * HS, NB, 4 * HS, HS, 1, stream);
        lstm_update<<<(nBH + EW - 1) / EW, EW, 0, stream>>>(
            gates, c_sa, hseq + (size_t)t * NB * HS, nullptr, nullptr);
    }

    // -------- temporal-attention LSTM scan --------
    for (int t = 0; t < S; ++t) {
        const float* ht = hseq + (size_t)t * NB * HS;
        // beta = softmax( tanh(h_t @ ta_Wa + ta_ba) @ ta_Va )   (s0 term == 0)
        launch_guard(ht, HS, ta_Wa, S, ta_ba, bpre, S, NB, S, HS, 0, stream);
        ew_tanh<<<(NB * S + EW - 1) / EW, EW, 0, stream>>>(bpre, NB * S);
        launch_guard(bpre, S, ta_Va, S, nullptr, beta, S, NB, S, S, 0, stream);
        ta_softmax16<<<(NB + EW - 1) / EW, EW, 0, stream>>>(beta);
        ta_context<<<(nBH + EW - 1) / EW, EW, 0, stream>>>(beta, hseq, hctx);
        // gates = h_ctx @ ta_W + lh @ ta_U + y_prev (x) ta_Wy + ta_b
        launch_full(hctx, HT, ta_WT, HS, ta_b,    gates, 4 * HT, NB, 4 * HT, HT, 0, stream);
        launch_full(lh,   HT, ta_UT, HT, nullptr, gates, 4 * HT, NB, 4 * HT, HT, 1, stream);
        lstm_update<<<(nBH + EW - 1) / EW, EW, 0, stream>>>(gates, lc, lh, yb, ta_Wy);
        ta_output<<<NB / 8, 256, 0, stream>>>(lh, fc_W, fc_b, yb);
    }

    copy_out<<<(NB + EW - 1) / EW, EW, 0, stream>>>(yb, out);
}
